// SkipTransformer_22892175688351
// MI455X (gfx1250) — compile-verified, matrix operand order checked by reading, not run on hardware
//
#include <hip/hip_runtime.h>
#include <hip/hip_bf16.h>

#define B_   4
#define N_   2048
#define K_   16
#define CIN_ 128
#define DIM_ 256
#define PH_  64
#define AH_  1024
#define POINTS 4
#define COLS (POINTS * K_) // 64 neighbor-columns per attention workgroup
#define ATTLD (COLS + 1)   // padded f32 LDS leading dim for scores

typedef __attribute__((ext_vector_type(16))) _Float16 v16h;
typedef __attribute__((ext_vector_type(8)))  _Float16 v8h;
typedef __attribute__((ext_vector_type(8)))  float    v8f;
typedef __attribute__((ext_vector_type(4)))  float    v4f;

__device__ __forceinline__ v8f wmma32(v16h a, v16h b, v8f c) {
  return __builtin_amdgcn_wmma_f32_16x16x32_f16(false, a, false, b, (short)0, c,
                                                false, false);
}

// Swizzled f16 LDS layout: element (k,col) of a K x (nct*16) matrix lives at
// blk*512 + lane*16 + vec*8 + sub, matching the WMMA B-fragment register
// order, so fragment loads are 2x ds_load_b128 and epilogue stores are 1x b128.
__device__ __forceinline__ int swzIdx(int k, int col, int nct) {
  int kb = k >> 5, kr = k & 31;
  int seg = kr >> 3, sub = kr & 7;
  return (((kb * nct + (col >> 4)) << 9) | ((((seg & 1) << 4) | (col & 15)) << 4) |
          ((seg >> 1) << 3) | sub);
}

// ---- fragment loaders (wave32, 16x16x32 f16 WMMA) ------------------------
// A (16x32, rows row0.., K-block k0..k0+31), f16 row-major, leading dim lda.
__device__ __forceinline__ v16h ldA(const _Float16* A, int lda, int row0, int k0) {
  int l = threadIdx.x & 31;
  const _Float16* p = A + (row0 + (l & 15)) * lda + k0 + ((l & 16) ? 8 : 0);
  v16h f;
#pragma unroll
  for (int i = 0; i < 8; ++i) f[i] = p[i];
#pragma unroll
  for (int i = 0; i < 8; ++i) f[8 + i] = p[16 + i];
  return f;
}
// B fragment from swizzled f16 LDS: contiguous 32B per lane.
__device__ __forceinline__ v16h ldBswz(const _Float16* Bm, int nct, int kb, int ct) {
  const _Float16* p = Bm + (((kb * nct + ct) << 9) | ((threadIdx.x & 31) << 4));
  return *(const v16h*)p;
}
// Stage one 8-row x 4-col f32 global sub-tile into swizzled f16 LDS.
__device__ __forceinline__ void stageOct(const float* rowbase, size_t ld, int k0,
                                         int cg, _Float16* dst, int nct, int ct) {
  v4f v[8];
#pragma unroll
  for (int j = 0; j < 8; ++j) v[j] = *(const v4f*)(rowbase + ld * j + cg * 4);
#pragma unroll
  for (int cc = 0; cc < 4; ++cc) {
    v8h o;
#pragma unroll
    for (int j = 0; j < 8; ++j) o[j] = (_Float16)v[j][cc];
    *(v8h*)(dst + swzIdx(k0, ct * 16 + cg * 4 + cc, nct)) = o;
  }
}

// ---- f16 weight arena layout (element offsets) ---------------------------
#define O_V1  0
#define S_V1  (CIN_ * 2 * CIN_)
#define O_V2  (O_V1 + S_V1)
#define S_V2  (CIN_ * CIN_)
#define O_VS  (O_V2 + S_V2)
#define S_VS  (CIN_ * 2 * CIN_)
#define O_K   (O_VS + S_VS)
#define S_KQ  (DIM_ * CIN_)
#define O_Q   (O_K + S_KQ)
#define O_VAL (O_Q + S_KQ)
#define O_P2  (O_VAL + S_KQ)
#define S_P2  (DIM_ * PH_)
#define O_A1  (O_P2 + S_P2)
#define S_A1  (AH_ * DIM_)
#define O_A2  (O_A1 + S_A1)
#define S_A2  (DIM_ * AH_)
#define O_END (O_A2 + S_A2)
#define S_END (CIN_ * DIM_)
#define HTOTAL (O_END + S_END)

// ---- prep: convert weights to f16 (fold BN scale into w_a1), fuse b_a1 ---
__global__ void prep_kernel(const float* w_v1, const float* w_v2, const float* w_vs,
                            const float* w_k, const float* w_q, const float* w_val,
                            const float* w_p2, const float* w_a1, const float* w_a2,
                            const float* w_end, const float* b_a1, const float* g_a,
                            const float* be_a, _Float16* W, float* ba1f) {
  int i = blockIdx.x * 256 + threadIdx.x;
  const float SBN = rsqrtf(1.0f + 1e-5f);
  if (i < AH_) ba1f[i] = b_a1[i] * g_a[i] * SBN + be_a[i];
  if (i >= HTOTAL) return;
  float v;
  if (i < O_V2)       v = w_v1[i - O_V1];
  else if (i < O_VS)  v = w_v2[i - O_V2];
  else if (i < O_K)   v = w_vs[i - O_VS];
  else if (i < O_Q)   v = w_k[i - O_K];
  else if (i < O_VAL) v = w_q[i - O_Q];
  else if (i < O_P2)  v = w_val[i - O_VAL];
  else if (i < O_A1)  v = w_p2[i - O_P2];
  else if (i < O_A2) { int j = i - O_A1; v = w_a1[j] * g_a[j / DIM_] * SBN; }
  else if (i < O_END) v = w_a2[i - O_A2];
  else                v = w_end[i - O_END];
  W[i] = (_Float16)v;
}

// ---- brute-force KNN: top-16 ascending (self included, dist 0) -----------
__global__ void knn_kernel(const float* pos, int* idx) {
  __shared__ float px[N_], py[N_], pz[N_];
  int b = blockIdx.y;
  const float* pb = pos + (size_t)b * 3 * N_;
  for (int i = threadIdx.x; i < N_; i += 256) {
    px[i] = pb[i]; py[i] = pb[N_ + i]; pz[i] = pb[2 * N_ + i];
  }
  __syncthreads();
  int n = blockIdx.x * 256 + threadIdx.x;
  float bd[K_]; int bi[K_];
#pragma unroll
  for (int j = 0; j < K_; ++j) { bd[j] = 3.402823e38f; bi[j] = 0; }
  float x = px[n], y = py[n], z = pz[n];
  for (int m = 0; m < N_; ++m) {
    float dx = x - px[m], dy = y - py[m], dz = z - pz[m];
    float d = dx * dx + dy * dy + dz * dz;
    if (d < bd[K_ - 1]) {
      int j = K_ - 1;
      while (j > 0 && bd[j - 1] > d) { bd[j] = bd[j - 1]; bi[j] = bi[j - 1]; --j; }
      bd[j] = d; bi[j] = m;
    }
  }
  int* o = idx + ((size_t)b * N_ + n) * K_;
  for (int j = 0; j < K_; ++j) o[j] = bi[j];
}

// ---- value = MLP_Res(cat(key,query)) : 256->128 relu ->128, + skip -------
__global__ void value_kernel(const float* key, const float* query, const _Float16* W,
                             const float* b_v1, const float* b_v2, const float* b_vs,
                             float* value) {
  __shared__ alignas(32) _Float16 xs[2 * CIN_ * 16];  // swizzled 256x16
  __shared__ alignas(32) _Float16 hid[CIN_ * 16];     // swizzled 128x16
  int b = blockIdx.x / (N_ / 16);
  int n0 = (blockIdx.x % (N_ / 16)) * 16;
  int tid = threadIdx.x, wave = tid >> 5;
  int l = tid & 31, c = l & 15, rh = (l & 16) ? 8 : 0;
  const float* keyb = key + (size_t)b * CIN_ * N_;
  const float* qryb = query + (size_t)b * CIN_ * N_;
  if (tid < 128) {  // 32 octets x 4 col-groups
    int oct = tid >> 2, cg = tid & 3;
    const float* rb = (oct < 16) ? keyb + (size_t)(oct * 8) * N_ + n0
                                 : qryb + (size_t)((oct - 16) * 8) * N_ + n0;
    stageOct(rb, N_, oct * 8, cg, xs, 1, 0);
  }
  __syncthreads();
  int o0 = wave * 16;
  v8f acc = {};
#pragma unroll
  for (int kb = 0; kb < 8; ++kb) {
    v16h a = ldA(W + O_V1, 2 * CIN_, o0, kb * 32);
    acc = wmma32(a, ldBswz(xs, 1, kb, 0), acc);
  }
  {
    v8h hv;
#pragma unroll
    for (int j = 0; j < 8; ++j) {
      float h = acc[j] + b_v1[o0 + rh + j];
      hv[j] = (_Float16)(h > 0.f ? h : 0.f);
    }
    *(v8h*)(hid + swzIdx(o0 + rh, c, 1)) = hv;
  }
  __syncthreads();
  v8f a2 = {};
#pragma unroll
  for (int kb = 0; kb < 4; ++kb) {
    v16h a = ldA(W + O_V2, CIN_, o0, kb * 32);
    a2 = wmma32(a, ldBswz(hid, 1, kb, 0), a2);
  }
#pragma unroll
  for (int kb = 0; kb < 8; ++kb) {
    v16h a = ldA(W + O_VS, 2 * CIN_, o0, kb * 32);
    a2 = wmma32(a, ldBswz(xs, 1, kb, 0), a2);
  }
  float* vb = value + (size_t)b * CIN_ * N_;
#pragma unroll
  for (int j = 0; j < 8; ++j) {
    int o = o0 + rh + j;
    vb[(size_t)o * N_ + n0 + c] = a2[j] + b_v2[o] + b_vs[o];
  }
}

// ---- k/q/v 1x1 convs (128->256): one block per (which,b,16-pt tile) ------
__global__ void kqv_kernel(const float* key, const float* query, const float* value,
                           const _Float16* W, const float* b_k, const float* b_q,
                           const float* b_val, float* kbuf, float* qbuf, float* vbuf) {
  __shared__ alignas(32) _Float16 ins[CIN_ * 16];  // swizzled 128x16
  int wb = blockIdx.x;
  int which = wb / (B_ * (N_ / 16));
  int rem = wb % (B_ * (N_ / 16));
  int b = rem / (N_ / 16);
  int n0 = (rem % (N_ / 16)) * 16;
  int tid = threadIdx.x, wave = tid >> 5;
  const _Float16* Wm = W + (which == 0 ? O_K : which == 1 ? O_Q : O_VAL);
  const float* bias = which == 0 ? b_k : which == 1 ? b_q : b_val;
  const float* in =
      (which == 0 ? key : which == 1 ? query : value) + (size_t)b * CIN_ * N_;
  float* out = (which == 0 ? kbuf : which == 1 ? qbuf : vbuf) + (size_t)b * DIM_ * N_;
  if (tid < 64) {  // 16 octets x 4 col-groups
    int oct = tid >> 2, cg = tid & 3;
    stageOct(in + (size_t)(oct * 8) * N_ + n0, N_, oct * 8, cg, ins, 1, 0);
  }
  __syncthreads();
  int l = tid & 31, c = l & 15, rh = (l & 16) ? 8 : 0;
  for (int ot = wave; ot < DIM_ / 16; ot += 8) {
    int o0 = ot * 16;
    v8f acc = {};
#pragma unroll
    for (int kb = 0; kb < 4; ++kb) {
      v16h a = ldA(Wm, CIN_, o0, kb * 32);
      acc = wmma32(a, ldBswz(ins, 1, kb, 0), acc);
    }
#pragma unroll
    for (int j = 0; j < 8; ++j) {
      int o = o0 + rh + j;
      out[(size_t)o * N_ + n0 + c] = acc[j] + bias[o];
    }
  }
}

// ---- fused attention: gather, pos_mlp, 256->1024->256, softmax, aggregate
__global__ void attn_kernel(const float* pos, const int* idx, const float* kbuf,
                            const float* qbuf, const float* vbuf, const _Float16* W,
                            const float* w_p1, const float* b_p1, const float* g_p,
                            const float* be_p, const float* b_p2, const float* ba1f,
                            const float* b_a2, float* agg) {
  extern __shared__ unsigned char smem[];
  float* att = (float*)smem;                                      // DIM_*ATTLD f32
  int* sidx = (int*)(smem + DIM_ * ATTLD * 4);                    // COLS ints
  _Float16* hp = (_Float16*)(smem + DIM_ * ATTLD * 4 + COLS * 4); // PH_*COLS swz
  _Float16* pe = hp + PH_ * COLS;                                 // DIM_*COLS swz
  _Float16* tt = pe + DIM_ * COLS;                                // DIM_*COLS swz
  _Float16* hh = tt + DIM_ * COLS;                                // AH_*COLS swz

  int tid = threadIdx.x, wave = tid >> 5;
  int l = tid & 31, cl = l & 15, rh = (l & 16) ? 8 : 0;
  int b = blockIdx.x / (N_ / POINTS);
  int base = (blockIdx.x % (N_ / POINTS)) * POINTS;
  const float* posb = pos + (size_t)b * 3 * N_;
  const float* kbb = kbuf + (size_t)b * DIM_ * N_;
  const float* qbb = qbuf + (size_t)b * DIM_ * N_;
  const float* vbb = vbuf + (size_t)b * DIM_ * N_;
  const float SBN = rsqrtf(1.0f + 1e-5f);

  // warm caches for the first weight tiles this wave will consume
  __builtin_prefetch(W + O_A1 + (size_t)(wave * 16) * DIM_, 0, 1);
  __builtin_prefetch(W + O_A2 + (size_t)(wave * 16) * AH_, 0, 1);

  if (tid < COLS) sidx[tid] = idx[((size_t)b * N_ + base) * K_ + tid];
  __syncthreads();

  // pos_mlp hidden: relu(bn(w_p1 . pos_rel + b_p1)); write one v8h per item
  for (int it = tid; it < (PH_ / 8) * COLS; it += 256) {
    int oct = it >> 6, col = it & 63;
    int p = col >> 4, m = sidx[col], n = base + p;
    float rx = posb[n] - posb[m];
    float ry = posb[N_ + n] - posb[N_ + m];
    float rz = posb[2 * N_ + n] - posb[2 * N_ + m];
    v8h hv;
#pragma unroll
    for (int j = 0; j < 8; ++j) {
      int o = oct * 8 + j;
      float h = w_p1[o * 3] * rx + w_p1[o * 3 + 1] * ry + w_p1[o * 3 + 2] * rz + b_p1[o];
      h = h * (g_p[o] * SBN) + be_p[o];
      hv[j] = (_Float16)(h > 0.f ? h : 0.f);
    }
    *(v8h*)(hp + swzIdx(oct * 8, col, POINTS)) = hv;
  }
  __syncthreads();

  // pe = w_p2 . hp + b_p2 ; t = (q - key_g) + pe
  for (int ot = wave; ot < DIM_ / 16; ot += 8) {
    int o0 = ot * 16;
    v8f acc[POINTS] = {};
#pragma unroll
    for (int kb = 0; kb < PH_ / 32; ++kb) {
      v16h a = ldA(W + O_P2, PH_, o0, kb * 32);
#pragma unroll
      for (int ct = 0; ct < POINTS; ++ct)
        acc[ct] = wmma32(a, ldBswz(hp, POINTS, kb, ct), acc[ct]);
    }
#pragma unroll
    for (int ct = 0; ct < POINTS; ++ct) {
      int col = ct * 16 + cl, n = base + ct, m = sidx[col];
      v8h pv, tv;
#pragma unroll
      for (int j = 0; j < 8; ++j) {
        int o = o0 + rh + j;
        float pev = acc[ct][j] + b_p2[o];
        pv[j] = (_Float16)pev;
        tv[j] = (_Float16)(qbb[(size_t)o * N_ + n] - kbb[(size_t)o * N_ + m] + pev);
      }
      *(v8h*)(pe + swzIdx(o0 + rh, col, POINTS)) = pv;
      *(v8h*)(tt + swzIdx(o0 + rh, col, POINTS)) = tv;
    }
  }
  __syncthreads();

  // h = relu(w_a1(bn-folded) . t + ba1f)
  for (int ot = wave; ot < AH_ / 16; ot += 8) {
    int o0 = ot * 16;
    v8f acc[POINTS] = {};
#pragma unroll
    for (int kb = 0; kb < DIM_ / 32; ++kb) {
      v16h a = ldA(W + O_A1, DIM_, o0, kb * 32);
#pragma unroll
      for (int ct = 0; ct < POINTS; ++ct)
        acc[ct] = wmma32(a, ldBswz(tt, POINTS, kb, ct), acc[ct]);
    }
#pragma unroll
    for (int ct = 0; ct < POINTS; ++ct) {
      int col = ct * 16 + cl;
      v8h hv;
#pragma unroll
      for (int j = 0; j < 8; ++j) {
        float h = acc[ct][j] + ba1f[o0 + rh + j];
        hv[j] = (_Float16)(h > 0.f ? h : 0.f);
      }
      *(v8h*)(hh + swzIdx(o0 + rh, col, POINTS)) = hv;
    }
  }
  __syncthreads();

  // att = w_a2 . h + b_a2
  for (int ot = wave; ot < DIM_ / 16; ot += 8) {
    int o0 = ot * 16;
    v8f acc[POINTS] = {};
#pragma unroll 8
    for (int kb = 0; kb < AH_ / 32; ++kb) {
      v16h a = ldA(W + O_A2, AH_, o0, kb * 32);
#pragma unroll
      for (int ct = 0; ct < POINTS; ++ct)
        acc[ct] = wmma32(a, ldBswz(hh, POINTS, kb, ct), acc[ct]);
    }
#pragma unroll
    for (int ct = 0; ct < POINTS; ++ct) {
      int col = ct * 16 + cl;
#pragma unroll
      for (int j = 0; j < 8; ++j) {
        int o = o0 + rh + j;
        att[o * ATTLD + col] = acc[ct][j] + b_a2[o];
      }
    }
  }
  __syncthreads();

  // softmax over K per (channel, point) + aggregate with (v + pe)
  for (int e = tid; e < DIM_ * POINTS; e += 256) {
    int c = e & (DIM_ - 1), p = e >> 8, n = base + p;
    const float* row = att + c * ATTLD + p * 16;
    float mx = row[0];
#pragma unroll
    for (int k = 1; k < 16; ++k) mx = fmaxf(mx, row[k]);
    float s = 0.f, ex[16];
#pragma unroll
    for (int k = 0; k < 16; ++k) { ex[k] = __expf(row[k] - mx); s += ex[k]; }
    float inv = 1.f / s;
    float vn = vbb[(size_t)c * N_ + n];
    float a = 0.f;
#pragma unroll
    for (int k = 0; k < 16; ++k)
      a += ex[k] * inv * (vn + (float)pe[swzIdx(c, p * 16 + k, POINTS)]);
    agg[(size_t)b * DIM_ * N_ + (size_t)c * N_ + n] = a;
  }
}

// ---- y = w_end . agg + b_end + identity(value) ---------------------------
__global__ void end_kernel(const float* agg, const float* value, const _Float16* W,
                           const float* b_end, float* out) {
  __shared__ alignas(32) _Float16 ag[DIM_ * 16];  // swizzled 256x16
  int b = blockIdx.x / (N_ / 16);
  int n0 = (blockIdx.x % (N_ / 16)) * 16;
  int tid = threadIdx.x, wave = tid >> 5;
  const float* ab = agg + (size_t)b * DIM_ * N_;
  if (tid < 128) {  // 32 octets x 4 col-groups
    int oct = tid >> 2, cg = tid & 3;
    stageOct(ab + (size_t)(oct * 8) * N_ + n0, N_, oct * 8, cg, ag, 1, 0);
  }
  __syncthreads();
  int o0 = wave * 16;
  v8f acc = {};
#pragma unroll
  for (int kb = 0; kb < DIM_ / 32; ++kb) {
    v16h a = ldA(W + O_END, DIM_, o0, kb * 32);
    acc = wmma32(a, ldBswz(ag, 1, kb, 0), acc);
  }
  int l = tid & 31, c = l & 15, rh = (l & 16) ? 8 : 0;
  const float* vb = value + (size_t)b * CIN_ * N_;
  float* ob = out + (size_t)b * CIN_ * N_;
#pragma unroll
  for (int j = 0; j < 8; ++j) {
    int o = o0 + rh + j;
    ob[(size_t)o * N_ + n0 + c] = acc[j] + b_end[o] + vb[(size_t)o * N_ + n0 + c];
  }
}

extern "C" void kernel_launch(void* const* d_in, const int* in_sizes, int n_in,
                              void* d_out, int out_size, void* d_ws, size_t ws_size,
                              hipStream_t stream) {
  const float* pos   = (const float*)d_in[0];
  const float* key   = (const float*)d_in[1];
  const float* query = (const float*)d_in[2];
  const float* w_v1  = (const float*)d_in[3];
  const float* b_v1  = (const float*)d_in[4];
  const float* w_v2  = (const float*)d_in[5];
  const float* b_v2  = (const float*)d_in[6];
  const float* w_vs  = (const float*)d_in[7];
  const float* b_vs  = (const float*)d_in[8];
  const float* w_k   = (const float*)d_in[9];
  const float* b_k   = (const float*)d_in[10];
  const float* w_q   = (const float*)d_in[11];
  const float* b_q   = (const float*)d_in[12];
  const float* w_val = (const float*)d_in[13];
  const float* b_val = (const float*)d_in[14];
  const float* w_p1  = (const float*)d_in[15];
  const float* b_p1  = (const float*)d_in[16];
  const float* g_p   = (const float*)d_in[17];
  const float* be_p  = (const float*)d_in[18];
  const float* w_p2  = (const float*)d_in[19];
  const float* b_p2  = (const float*)d_in[20];
  const float* w_a1  = (const float*)d_in[21];
  const float* b_a1  = (const float*)d_in[22];
  const float* g_a   = (const float*)d_in[23];
  const float* be_a  = (const float*)d_in[24];
  const float* w_a2  = (const float*)d_in[25];
  const float* b_a2  = (const float*)d_in[26];
  const float* w_end = (const float*)d_in[27];
  const float* b_end = (const float*)d_in[28];

  float* ws = (float*)d_ws;
  float* valuef = ws;  ws += (size_t)B_ * CIN_ * N_;
  float* kbuf = ws;    ws += (size_t)B_ * DIM_ * N_;
  float* qbuf = ws;    ws += (size_t)B_ * DIM_ * N_;
  float* vbuf = ws;    ws += (size_t)B_ * DIM_ * N_;
  float* aggf = ws;    ws += (size_t)B_ * DIM_ * N_;
  float* ba1f = ws;    ws += AH_;
  int* idxb = (int*)ws; ws += (size_t)B_ * N_ * K_;
  _Float16* Wh = (_Float16*)ws;

  prep_kernel<<<(HTOTAL + 255) / 256, 256, 0, stream>>>(
      w_v1, w_v2, w_vs, w_k, w_q, w_val, w_p2, w_a1, w_a2, w_end,
      b_a1, g_a, be_a, Wh, ba1f);
  knn_kernel<<<dim3(N_ / 256, B_), 256, 0, stream>>>(pos, idxb);
  value_kernel<<<B_ * (N_ / 16), 256, 0, stream>>>(key, query, Wh, b_v1, b_v2, b_vs,
                                                   valuef);
  kqv_kernel<<<3 * B_ * (N_ / 16), 256, 0, stream>>>(
      key, query, valuef, Wh, b_k, b_q, b_val, kbuf, qbuf, vbuf);
  size_t smem = (size_t)DIM_ * ATTLD * 4 + COLS * 4 +
                (size_t)(PH_ + DIM_ + DIM_ + AH_) * COLS * 2;
  attn_kernel<<<B_ * (N_ / POINTS), 256, smem, stream>>>(
      pos, idxb, kbuf, qbuf, vbuf, Wh, w_p1, b_p1, g_p, be_p, b_p2, ba1f, b_a2, aggf);
  end_kernel<<<B_ * (N_ / 16), 256, 0, stream>>>(aggf, valuef, Wh, b_end,
                                                 (float*)d_out);
}
